// MambaAttentionChunkDouble_25744033972438
// MI455X (gfx1250) — matrix operand order repaired and checked by reading, not compiled, not run on hardware
//
#include <hip/hip_runtime.h>
#include <hip/hip_bf16.h>
#include <math.h>

// Problem constants (B,H,W,C,N,D) = (16,64,64,256,8,32)
#define BB 16
#define HH 64
#define WWD 64
#define CC 256
#define NN 8
#define DD 32
#define PIX 4096          // HH*WWD
#define ETOT ((size_t)16777216)  // B*H*W*C elements

typedef __attribute__((ext_vector_type(16))) __bf16 bf16x16;
typedef __attribute__((ext_vector_type(8)))  float  floatx8;

union FragB16 { bf16x16 v; unsigned short u[16]; };
union FragF32 { floatx8 v; float f[8]; };

__device__ __forceinline__ unsigned short f2bf(float f) {
  unsigned int u = __float_as_uint(f);
  u += 0x7fffu + ((u >> 16) & 1u);           // round-to-nearest-even
  return (unsigned short)(u >> 16);
}
__device__ __forceinline__ float bf2f(unsigned short h) {
  return __uint_as_float(((unsigned int)h) << 16);
}

__device__ __forceinline__ floatx8 wmma_bf16(bf16x16 a, bf16x16 b, floatx8 c) {
  return __builtin_amdgcn_wmma_f32_16x16x32_bf16(false, a, false, b, (short)0, c, false, false);
}

// ---- WMMA fragment loaders (contiguous k order shared by A and B, wave32) ----
// A (16x32 bf16, M x K): lane&15 = M; lanes<16 hold K {0..15}, lanes>=16 hold K {16..31}
__device__ __forceinline__ bf16x16 load_A_frag(const unsigned short* s, int row_base, int col_base, int ld) {
  const int lane = threadIdx.x & 31;
  const int m = lane & 15;
  const int lo = (lane & 16) ? 16 : 0;
  const unsigned short* rp = s + (row_base + m) * ld + col_base;
  FragB16 a;
#pragma unroll
  for (int i = 0; i < 8; ++i) { a.u[i] = rp[lo + i]; a.u[8 + i] = rp[8 + lo + i]; }
  return a.v;
}
// B (32x16 bf16, K x N) where B[k][n] = src[n][k] (src row-major, rows = N dim): contiguous 16 along k
__device__ __forceinline__ bf16x16 load_Bt_frag(const unsigned short* s, int n_base, int col_base, int ld) {
  const int lane = threadIdx.x & 31;
  const int n = lane & 15;
  const int kb = (lane & 16) ? 16 : 0;
  const unsigned short* rp = s + (n_base + n) * ld + col_base + kb;
  FragB16 b;
#pragma unroll
  for (int i = 0; i < 16; ++i) b.u[i] = rp[i];
  return b.v;
}
// B (32x16) where B[k][n] = src[k][n] (src row-major, rows = K dim)
__device__ __forceinline__ bf16x16 load_Bd_frag(const unsigned short* s, int k_base, int n_base, int ld) {
  const int lane = threadIdx.x & 31;
  const int n = lane & 15;
  const int kb = (lane & 16) ? 16 : 0;
  FragB16 b;
#pragma unroll
  for (int i = 0; i < 16; ++i) b.u[i] = s[(k_base + kb + i) * ld + n_base + n];
  return b.v;
}

// ---------------- 65536x256x256 projection GEMM:  out = A @ W^T + bias ----------------
// mode 0: q -> qk layout [b,n,h,w,d] bf16 ; mode 1: k (scaled) same ; mode 2: v -> [b,h,w,c] bf16 ; mode 3: fp32 row-major out
__global__ __launch_bounds__(128)
void gemm256_kernel(const float* __restrict__ Ain, const float* __restrict__ Wmat,
                    const float* __restrict__ bias, unsigned short* __restrict__ qkout,
                    unsigned short* __restrict__ vout, float* __restrict__ fout, int mode) {
  __shared__ alignas(16) unsigned short As[64 * 32];
  __shared__ alignas(16) unsigned short Bs[64 * 32];
  const int tid = threadIdx.x;
  const int lane = tid & 31;
  const int wv = tid >> 5;
  const int rowBase = blockIdx.x * 64;
  const int colBase = blockIdx.y * 64;

  FragF32 acc[4];
#pragma unroll
  for (int jt = 0; jt < 4; ++jt)
#pragma unroll
    for (int r = 0; r < 8; ++r) acc[jt].f[r] = 0.0f;

  for (int kk = 0; kk < 256; kk += 32) {
    __syncthreads();
    for (int idx = tid; idx < 64 * 8; idx += 128) {        // stage A tile (64x32) as bf16
      int row = idx >> 3, c4 = idx & 7;
      const float* sp = Ain + (size_t)(rowBase + row) * 256 + kk + c4 * 4;
      unsigned short* dp = As + row * 32 + c4 * 4;
      dp[0] = f2bf(sp[0]); dp[1] = f2bf(sp[1]); dp[2] = f2bf(sp[2]); dp[3] = f2bf(sp[3]);
    }
    for (int idx = tid; idx < 64 * 8; idx += 128) {        // stage W tile (64 out-ch x 32 k)
      int row = idx >> 3, c4 = idx & 7;
      const float* sp = Wmat + (size_t)(colBase + row) * 256 + kk + c4 * 4;
      unsigned short* dp = Bs + row * 32 + c4 * 4;
      dp[0] = f2bf(sp[0]); dp[1] = f2bf(sp[1]); dp[2] = f2bf(sp[2]); dp[3] = f2bf(sp[3]);
    }
    __syncthreads();
    bf16x16 afrag = load_A_frag(As, wv * 16, 0, 32);
#pragma unroll
    for (int jt = 0; jt < 4; ++jt) {
      bf16x16 bfrag = load_Bt_frag(Bs, jt * 16, 0, 32);
      acc[jt].v = wmma_bf16(afrag, bfrag, acc[jt].v);
    }
  }
  const int rowOff = (lane & 16) ? 8 : 0;
#pragma unroll
  for (int jt = 0; jt < 4; ++jt) {
    int col = colBase + jt * 16 + (lane & 15);
    float bcol = bias[col];
#pragma unroll
    for (int r = 0; r < 8; ++r) {
      int p = rowBase + wv * 16 + r + rowOff;
      float val = acc[jt].f[r] + bcol;
      if (mode == 1) val *= 0.17677669529663687f;   // D^-0.5
      if (mode <= 1) {
        int n = col >> 5, d = col & 31, b = p >> 12;
        qkout[((size_t)(b * NN + n) * PIX + (p & (PIX - 1))) * DD + d] = f2bf(val);
      } else if (mode == 2) {
        vout[(size_t)p * CC + col] = f2bf(val);
      } else {
        fout[(size_t)p * CC + col] = val;
      }
    }
  }
}

// ---------------- decay + inclusive scan (segsum inputs) ----------------
// along_w=1: fixed h, scan over w, dt_w row 0 -> cs_a[b,n,h,w]; along_w=0: fixed w, scan over h, row 1 -> cs_b[b,n,w,h]
__global__ __launch_bounds__(64)
void decay_scan_kernel(const float* __restrict__ x, const float* __restrict__ dt_w,
                       const float* __restrict__ dt_bias, const float* __restrict__ A_log,
                       float* __restrict__ cs, int along_w) {
  __shared__ float s[64];
  const int bx = blockIdx.x;
  const int t = threadIdx.x;
  const int fix = bx & 63;
  const int n = (bx >> 6) & 7;
  const int b = bx >> 9;
  const int h = along_w ? fix : t;
  const int w = along_w ? t : fix;
  const float* xp = x + ((size_t)(b * HH + h) * WWD + w) * CC + n * DD;
  const float* wp = dt_w + (along_w ? 0 : DD);
  float dt = 0.f;
#pragma unroll
  for (int d = 0; d < DD; ++d) dt += xp[d] * wp[d];
  dt += dt_bias[n];
  float sp = dt > 20.f ? dt : log1pf(expf(dt));
  s[t] = sp * (-expf(A_log[n]));
  __syncthreads();
  for (int off = 1; off < 64; off <<= 1) {
    float add = (t >= off) ? s[t - off] : 0.f;
    __syncthreads();
    s[t] += add;
    __syncthreads();
  }
  cs[(size_t)bx * 64 + t] = s[t];
}

// ---------------- in-place RoPE (theta_shift) on q and k (bf16, [b,n,h,w,d]) ----------------
__global__ __launch_bounds__(256)
void rope_kernel(unsigned short* __restrict__ q, unsigned short* __restrict__ k,
                 const float* __restrict__ sinp, const float* __restrict__ cosp) {
  int t = blockIdx.x * 256 + threadIdx.x;  // 8,388,608 pair slots
  int dp = t & 15;
  int w = (t >> 4) & 63;
  int h = (t >> 10) & 63;
  int n = (t >> 16) & 7;
  int b = t >> 19;
  size_t base = (((size_t)(b * NN + n) * HH + h) * WWD + w) * DD + 2 * dp;
  int si = (h * WWD + w) * DD + 2 * dp;
  float c0 = cosp[si], c1 = cosp[si + 1], s0 = sinp[si], s1 = sinp[si + 1];
  float q0 = bf2f(q[base]), q1 = bf2f(q[base + 1]);
  q[base]     = f2bf(q0 * c0 - q1 * s0);
  q[base + 1] = f2bf(q1 * c1 + q0 * s1);
  float k0 = bf2f(k[base]), k1 = bf2f(k[base + 1]);
  k[base]     = f2bf(k0 * c0 - k1 * s0);
  k[base + 1] = f2bf(k1 * c1 + k0 * s1);
}

// ---------------- LePE depthwise 5x5 conv -> initializes fp32 staging buffer ----------------
__global__ __launch_bounds__(256)
void lepe_kernel(const unsigned short* __restrict__ v, const float* __restrict__ lw,
                 const float* __restrict__ lb, float* __restrict__ outst) {
  int g = blockIdx.x * 256 + threadIdx.x;  // 16,777,216
  int c = g & 255;
  int w = (g >> 8) & 63;
  int h = (g >> 14) & 63;
  int b = g >> 20;
  float acc = lb[c];
#pragma unroll
  for (int di = 0; di < 5; ++di) {
    int hh2 = h + di - 2;
    if (hh2 < 0 || hh2 >= HH) continue;
#pragma unroll
    for (int dj = 0; dj < 5; ++dj) {
      int ww2 = w + dj - 2;
      if (ww2 < 0 || ww2 >= WWD) continue;
      acc += bf2f(v[((size_t)(b * HH + hh2) * WWD + ww2) * CC + c]) * lw[(di * 5 + dj) * CC + c];
    }
  }
  outst[g] = acc;
}

// ---------------- attention passes (64x64 tile, masked softmax, WMMA) ----------------
// MODE 0: W-pass1 (fix=h): S=QK^T+mask_w, softmax, v_w = P@V  -> outb16 = vw [b,n,w,h,d]
// MODE 1: H-pass  (fix=x): S over h + mask_h, softmax; out1 = P@vw-slice -> 0.5 accum into outst;
//                           vh2 = P@V -> outb16 = vh2 [b,n,h,w,d]
// MODE 2: W-pass2 (fix=y): S over w + mask_w, softmax; out2 = P@vh2-slice -> 0.5 accum into outst
template <int MODE>
__global__ __launch_bounds__(128)
void attn_kernel(const unsigned short* __restrict__ qr, const unsigned short* __restrict__ kr,
                 const unsigned short* __restrict__ vbf, const unsigned short* __restrict__ vaux,
                 const float* __restrict__ cs, unsigned short* __restrict__ outb16,
                 float* __restrict__ outst) {
  __shared__ alignas(16) unsigned short Qs[64 * 32];
  __shared__ alignas(16) unsigned short Ks[64 * 32];
  __shared__ alignas(16) unsigned short Vs[64 * 32];
  __shared__ alignas(16) unsigned short Vs2[64 * 32];
  __shared__ alignas(16) unsigned short Ps[64 * 64];
  __shared__ float css[64];
  const int tid = threadIdx.x, lane = tid & 31, wv = tid >> 5;
  const int bx = blockIdx.x;
  const int fix = bx & 63;
  const int n = (bx >> 6) & 7;
  const int b = bx >> 9;

  {  // stage Q/K (64 rows x 32)
    size_t qbase; size_t rstride;
    if (MODE == 1) { qbase = (((size_t)(b * NN + n) * HH) * WWD + fix) * DD; rstride = (size_t)WWD * DD; }
    else           { qbase = (((size_t)(b * NN + n) * HH + fix) * WWD) * DD; rstride = DD; }
    for (int idx = tid; idx < 64 * 4; idx += 128) {
      int row = idx >> 2, c4 = idx & 3;
      *(uint4*)(Qs + row * 32 + c4 * 8) = *(const uint4*)(qr + qbase + row * rstride + c4 * 8);
      *(uint4*)(Ks + row * 32 + c4 * 8) = *(const uint4*)(kr + qbase + row * rstride + c4 * 8);
    }
  }
  for (int idx = tid; idx < 64 * 4; idx += 128) {  // stage V (and vw slice for MODE 1)
    int row = idx >> 2, c4 = idx & 3;
    if (MODE == 0) {
      *(uint4*)(Vs + row * 32 + c4 * 8) =
          *(const uint4*)(vbf + ((size_t)(b * HH + fix) * WWD + row) * CC + n * DD + c4 * 8);
    } else if (MODE == 1) {
      *(uint4*)(Vs + row * 32 + c4 * 8) =
          *(const uint4*)(vbf + ((size_t)(b * HH + row) * WWD + fix) * CC + n * DD + c4 * 8);
      *(uint4*)(Vs2 + row * 32 + c4 * 8) =
          *(const uint4*)(vaux + (((size_t)(b * NN + n) * WWD + fix) * HH + row) * DD + c4 * 8);
    } else {
      *(uint4*)(Vs + row * 32 + c4 * 8) =
          *(const uint4*)(vaux + (((size_t)(b * NN + n) * HH + fix) * WWD + row) * DD + c4 * 8);
    }
  }
  if (tid < 64) css[tid] = cs[(size_t)bx * 64 + tid];
  __syncthreads();

  // S = Q K^T
  FragF32 sacc[4];
  {
    bf16x16 afrag = load_A_frag(Qs, wv * 16, 0, 32);
#pragma unroll
    for (int jt = 0; jt < 4; ++jt) {
      FragF32 z;
#pragma unroll
      for (int r = 0; r < 8; ++r) z.f[r] = 0.f;
      bf16x16 bfrag = load_Bt_frag(Ks, jt * 16, 0, 32);
      sacc[jt].v = wmma_bf16(afrag, bfrag, z.v);
    }
  }
  // mask + softmax (row spread across 16 lanes x 4 accumulators)
  const int rowOff = (lane & 16) ? 8 : 0;
#pragma unroll
  for (int r = 0; r < 8; ++r) {
    int i = wv * 16 + r + rowOff;
    float ci = css[i];
    float sv[4], vmax = -3.4e38f;
#pragma unroll
    for (int jt = 0; jt < 4; ++jt) {
      int j = jt * 16 + (lane & 15);
      float cj = css[j];
      float mask = (i >= j) ? (ci - cj) : (cj - ci);   // segsum mask
      float tv = sacc[jt].f[r] + mask;
      sv[jt] = tv;
      vmax = fmaxf(vmax, tv);
    }
#pragma unroll
    for (int m = 1; m < 16; m <<= 1) vmax = fmaxf(vmax, __shfl_xor(vmax, m, 32));
    float sum = 0.f;
#pragma unroll
    for (int jt = 0; jt < 4; ++jt) { sv[jt] = expf(sv[jt] - vmax); sum += sv[jt]; }
#pragma unroll
    for (int m = 1; m < 16; m <<= 1) sum += __shfl_xor(sum, m, 32);
    float inv = 1.0f / sum;
#pragma unroll
    for (int jt = 0; jt < 4; ++jt) Ps[i * 64 + jt * 16 + (lane & 15)] = f2bf(sv[jt] * inv);
  }
  __syncthreads();

  // O = P @ V   (MODE 1 uses the vw slice here for out1)
#pragma unroll
  for (int nt = 0; nt < 2; ++nt) {
    FragF32 o;
#pragma unroll
    for (int r = 0; r < 8; ++r) o.f[r] = 0.f;
#pragma unroll
    for (int kc = 0; kc < 2; ++kc) {
      bf16x16 a = load_A_frag(Ps, wv * 16, kc * 32, 64);
      bf16x16 bb = load_Bd_frag((MODE == 1) ? Vs2 : Vs, kc * 32, nt * 16, 32);
      o.v = wmma_bf16(a, bb, o.v);
    }
#pragma unroll
    for (int r = 0; r < 8; ++r) {
      int i = wv * 16 + r + rowOff;
      int d = nt * 16 + (lane & 15);
      float val = o.f[r];
      if (MODE == 0) {
        outb16[(((size_t)(b * NN + n) * WWD + i) * HH + fix) * DD + d] = f2bf(val);  // vw [b,n,w,h,d]
      } else if (MODE == 1) {
        outst[((size_t)(b * HH + i) * WWD + fix) * CC + n * DD + d] += 0.5f * val;   // 0.5*out1
      } else {
        outst[((size_t)(b * HH + fix) * WWD + i) * CC + n * DD + d] += 0.5f * val;   // 0.5*out2
      }
    }
  }
  if (MODE == 1) {  // vh2 = P @ V
#pragma unroll
    for (int nt = 0; nt < 2; ++nt) {
      FragF32 o;
#pragma unroll
      for (int r = 0; r < 8; ++r) o.f[r] = 0.f;
#pragma unroll
      for (int kc = 0; kc < 2; ++kc) {
        bf16x16 a = load_A_frag(Ps, wv * 16, kc * 32, 64);
        bf16x16 bb = load_Bd_frag(Vs, kc * 32, nt * 16, 32);
        o.v = wmma_bf16(a, bb, o.v);
      }
#pragma unroll
      for (int r = 0; r < 8; ++r) {
        int i = wv * 16 + r + rowOff;
        int d = nt * 16 + (lane & 15);
        outb16[(((size_t)(b * NN + n) * HH + i) * WWD + fix) * DD + d] = f2bf(o.f[r]);  // vh2 [b,n,h,w,d]
      }
    }
  }
}

extern "C" void kernel_launch(void* const* d_in, const int* in_sizes, int n_in,
                              void* d_out, int out_size, void* d_ws, size_t ws_size,
                              hipStream_t stream) {
  const float* x       = (const float*)d_in[0];
  const float* sinp    = (const float*)d_in[1];
  const float* cosp    = (const float*)d_in[2];
  const float* Wq      = (const float*)d_in[3];
  const float* bq      = (const float*)d_in[4];
  const float* Wk      = (const float*)d_in[5];
  const float* bk      = (const float*)d_in[6];
  const float* Wv      = (const float*)d_in[7];
  const float* bv      = (const float*)d_in[8];
  const float* Wo      = (const float*)d_in[9];
  const float* bo      = (const float*)d_in[10];
  const float* lepe_w  = (const float*)d_in[11];
  const float* lepe_b  = (const float*)d_in[12];
  const float* dt_w    = (const float*)d_in[13];
  const float* dt_bias = (const float*)d_in[14];
  const float* A_log   = (const float*)d_in[15];
  float* out = (float*)d_out;

  // workspace carve (total ~239 MB)
  char* ws = (char*)d_ws;
  unsigned short* qrb  = (unsigned short*)ws; ws += ETOT * 2;   // q (rotated in-place)  [b,n,h,w,d]
  unsigned short* krb  = (unsigned short*)ws; ws += ETOT * 2;   // k (rotated in-place)
  unsigned short* vb   = (unsigned short*)ws; ws += ETOT * 2;   // v  [b,h,w,c]
  unsigned short* vwb  = (unsigned short*)ws; ws += ETOT * 2;   // v_w  [b,n,w,h,d]
  unsigned short* vh2b = (unsigned short*)ws; ws += ETOT * 2;   // v_h2 [b,n,h,w,d]
  float* cs_a  = (float*)ws; ws += (size_t)BB * NN * HH * WWD * 4;
  float* cs_b  = (float*)ws; ws += (size_t)BB * NN * HH * WWD * 4;
  float* outst = (float*)ws; ws += ETOT * 4;                    // lepe + 0.5*out1 + 0.5*out2

  dim3 gblk(128);
  dim3 ggrid(1024, 4);
  gemm256_kernel<<<ggrid, gblk, 0, stream>>>(x, Wq, bq, qrb, nullptr, nullptr, 0);
  gemm256_kernel<<<ggrid, gblk, 0, stream>>>(x, Wk, bk, krb, nullptr, nullptr, 1);
  gemm256_kernel<<<ggrid, gblk, 0, stream>>>(x, Wv, bv, nullptr, vb, nullptr, 2);

  decay_scan_kernel<<<BB * NN * HH, 64, 0, stream>>>(x, dt_w, dt_bias, A_log, cs_a, 1);
  decay_scan_kernel<<<BB * NN * WWD, 64, 0, stream>>>(x, dt_w, dt_bias, A_log, cs_b, 0);

  rope_kernel<<<32768, 256, 0, stream>>>(qrb, krb, sinp, cosp);
  lepe_kernel<<<65536, 256, 0, stream>>>(vb, lepe_w, lepe_b, outst);

  attn_kernel<0><<<BB * NN * HH, 128, 0, stream>>>(qrb, krb, vb, nullptr, cs_a, vwb, nullptr);
  attn_kernel<1><<<BB * NN * WWD, 128, 0, stream>>>(qrb, krb, vb, vwb, cs_b, vh2b, outst);
  attn_kernel<2><<<BB * NN * HH, 128, 0, stream>>>(qrb, krb, vb, vh2b, cs_a, nullptr, outst);

  gemm256_kernel<<<ggrid, gblk, 0, stream>>>(outst, Wo, bo, nullptr, nullptr, out, 3);
}
